// Encoder_31104153157725
// MI455X (gfx1250) — matrix operand order repaired and checked by reading, not compiled, not run on hardware
//
#include <hip/hip_runtime.h>
#include <hip/hip_bf16.h>
#include <stdint.h>

#define K_KERNELS 4
#define C_OUT     128
#define DIN0      256
#define DEPTH     3
#define ATTN_SLOPE 0.2f
#define ACT_SLOPE  0.01f
#define COLS_PER_BLOCK 64

#define HAVE_TDM __has_builtin(__builtin_amdgcn_tensor_load_to_lds)

typedef __attribute__((ext_vector_type(16))) __bf16 v16bf;
typedef __attribute__((ext_vector_type(8)))  float  v8f;
typedef __attribute__((ext_vector_type(4)))  unsigned int u32x4;
typedef __attribute__((ext_vector_type(8)))  int i32x8;
typedef __attribute__((ext_vector_type(4)))  int i32x4;

union Frag { v16bf v; uint4 q[2]; unsigned short u[16]; };

__device__ __forceinline__ unsigned short f2bf(float f) {
    unsigned u = __float_as_uint(f);
    u += 0x7FFFu + ((u >> 16) & 1u);     // round-to-nearest-even
    return (unsigned short)(u >> 16);
}
__device__ __forceinline__ unsigned encOrd(float f) {  // order-preserving uint encode
    unsigned u = __float_as_uint(f);
    return (u & 0x80000000u) ? ~u : (u | 0x80000000u);
}
__device__ __forceinline__ float decOrd(unsigned u) {
    unsigned v = (u & 0x80000000u) ? (u & 0x7FFFFFFFu) : ~u;
    return __uint_as_float(v);
}
__device__ __forceinline__ int binOf(float a) {
    int b = (int)(a * 256.0f);           // attn in [0, 4] -> 1024 bins
    if (b < 0) b = 0; if (b > 1023) b = 1023;
    return b;
}

// ---------------- conversions ----------------
__global__ void cvt_f32_bf16(const float* __restrict__ in, unsigned short* __restrict__ out, int n) {
    int i = blockIdx.x * blockDim.x + threadIdx.x;
    if (i < n) out[i] = f2bf(in[i]);
}
// W [nk][Din][C_OUT] f32  ->  wbT [nk][C_OUT][Din] bf16  (K-contiguous per column)
__global__ void cvt_w_transpose(const float* __restrict__ W, unsigned short* __restrict__ wbT,
                                int Din, int nk) {
    int i = blockIdx.x * blockDim.x + threadIdx.x;
    int total = nk * C_OUT * Din;
    if (i >= total) return;
    int k  = i / (C_OUT * Din);
    int r  = i - k * C_OUT * Din;
    int n  = r / Din;
    int kk = r - n * Din;
    wbT[i] = f2bf(W[((size_t)k * Din + kk) * C_OUT + n]);
}

// ---------------- WMMA GEMM: h[k] = xb @ W[k],  [N,Din]x[Din,128] ----------------
// grid: (ceil(N/128), 2 col-groups of 64, 4 kernels), block 256 = 8 waves.
// Each wave owns a 16x64 output strip: one A fragment feeds 4 WMMAs.
// B tile is [64 cols][Din K] in LDS (K-contiguous per lane), padded +8 ushorts/row
// for bank-conflict-free ds_load_b128; filled by the Tensor Data Mover when available.
__launch_bounds__(256)
__global__ void gemm_wmma_kernel(const unsigned short* __restrict__ xb,
                                 const unsigned short* __restrict__ wbT,
                                 float* __restrict__ h, int N, int Din) {
    __shared__ unsigned short Bl[16896];               // 64 * (256+8) max
    const int k    = blockIdx.z;
    const int col0 = blockIdx.y * COLS_PER_BLOCK;
    const int tid  = threadIdx.x;
    const unsigned short* wsrc = wbT + ((size_t)k * C_OUT + col0) * Din;

#if HAVE_TDM
    const int bstride = Din + 8;                       // TDM pad: +4 DWORDs per row
    if (tid < 32) {
        unsigned long long ga = (unsigned long long)(const void*)wsrc;
        const int padI = (Din == 256) ? 6 : 5;         // pad_interval: 128 / 64 DWORDs (one row)
        u32x4 g0;
        g0.x = 1u;                                          // count=1, is_restore=0
        g0.y = 0u;                                          // lds_addr = base of Bl
        g0.z = (unsigned)(ga & 0xFFFFFFFFu);                // global_addr[31:0]
        g0.w = (unsigned)((ga >> 32) & 0x1FFFFFFu) | (2u << 30);  // global_addr[56:32], type=2
        i32x8 g1;
        g1[0] = (1 << 16) | (1 << 20) | (padI << 22) | (3 << 25); // data_size=2B, pad_en, pad=4 DW
        g1[1] = (Din & 0xFFFF) << 16;                       // tensor_dim0 = Din (x contiguous = K)
        g1[2] = ((Din >> 16) & 0xFFFF) | ((C_OUT & 0xFFFF) << 16);  // dim0 hi | tensor_dim1 = C
        g1[3] = ((C_OUT >> 16) & 0xFFFF) | ((Din & 0xFFFF) << 16);  // dim1 hi | tile_dim0 = Din
        g1[4] = COLS_PER_BLOCK;                             // tile_dim1 = 64 cols, tile_dim2 = 0
        g1[5] = Din;                                        // tensor_dim0_stride = Din
        g1[6] = 0;
        g1[7] = 0;
        i32x4 z4 = {0, 0, 0, 0};
        i32x8 z8 = {0, 0, 0, 0, 0, 0, 0, 0};
        __builtin_amdgcn_tensor_load_to_lds(g0, g1, z4, z4, z8, 0);
        __builtin_amdgcn_s_wait_tensorcnt(0);
    }
    __syncthreads();
    { // escape Bl so the compiler knows the TDM wrote LDS (prevents undef-folding the reads)
        const unsigned short* esc = &Bl[0];
        asm volatile("" :: "v"(esc) : "memory");
    }
#else
    const int bstride = 264;
    const int dshift = (Din == 256) ? 8 : 7;
    for (int e = tid; e < COLS_PER_BLOCK * Din; e += 256) {
        Bl[(e >> dshift) * 264 + (e & (Din - 1))] = wsrc[e];
    }
    __syncthreads();
#endif

    const int wave = tid >> 5, lane = tid & 31;
    const int row0 = blockIdx.x * 128 + wave * 16;
    const int hlf  = lane >> 4;                        // 0/1 lane half
    const int l16  = lane & 15;
    int arow = row0 + l16; if (arow >= N) arow = N - 1;
    const unsigned short* aptr = xb + (size_t)arow * Din;
    v8f acc[4] = {};
    for (int k0 = 0; k0 < Din; k0 += 32) {
        Frag a;
        const int ak = k0 + hlf * 8;                   // A 16x32 bf16 layout (ISA 7.12.2)
        a.q[0] = *reinterpret_cast<const uint4*>(aptr + ak);
        a.q[1] = *reinterpret_cast<const uint4*>(aptr + ak + 16);
        __builtin_prefetch(aptr + ak + 32, 0, 1);      // global_prefetch_b8
        const int bk = k0 + hlf * 8;                   // B 32x16 bf16: same K layout as A
        // Load all four B fragments into independent registers first so the
        // scheduler can clause the 8 ds_load_b128 and run 4 WMMAs back-to-back.
        Frag b[4];
#pragma unroll
        for (int t = 0; t < 4; ++t) {
            const unsigned short* bp = Bl + (t * 16 + l16) * bstride + bk;
            b[t].q[0] = *reinterpret_cast<const uint4*>(bp);
            b[t].q[1] = *reinterpret_cast<const uint4*>(bp + 16);
        }
#pragma unroll
        for (int t = 0; t < 4; ++t) {
            acc[t] = __builtin_amdgcn_wmma_f32_16x16x32_bf16(false, a.v, false, b[t].v,
                                                             (short)0, acc[t], false, false);
        }
    }
    float* hk = h + (size_t)k * N * C_OUT;
#pragma unroll
    for (int t = 0; t < 4; ++t) {
        for (int r = 0; r < 8; ++r) {
            int row = row0 + hlf * 8 + r;              // C/D f32 layout (ISA 7.12.2)
            if (row < N) hk[(size_t)row * C_OUT + col0 + t * 16 + l16] = acc[t][r];
        }
    }
}

// ---------------- per-node attention scores: es/ed[k][n] = h[k][n] . a ----------------
__launch_bounds__(256)
__global__ void score_kernel(const float* __restrict__ h,
                             const float* __restrict__ a_s, const float* __restrict__ a_d,
                             float* __restrict__ es, float* __restrict__ ed, int N) {
    int wave = (blockIdx.x * blockDim.x + threadIdx.x) >> 5;
    int lane = threadIdx.x & 31;
    if (wave >= N) return;
    for (int k = 0; k < K_KERNELS; ++k) {
        const float* hr = h + ((size_t)k * N + wave) * C_OUT;
        float ss = 0.f, sd = 0.f;
        for (int c = lane; c < C_OUT; c += 32) {
            float v = hr[c];
            ss += v * a_s[k * C_OUT + c];
            sd += v * a_d[k * C_OUT + c];
        }
        for (int off = 16; off; off >>= 1) {
            ss += __shfl_xor(ss, off, 32);
            sd += __shfl_xor(sd, off, 32);
        }
        if (lane == 0) { es[(size_t)k * N + wave] = ss; ed[(size_t)k * N + wave] = sd; }
    }
}

// ---------------- edge list setup ----------------
__global__ void init_edges(const long long* __restrict__ ei, int* __restrict__ src,
                           int* __restrict__ dst, int E, int N) {
    int j = blockIdx.x * blockDim.x + threadIdx.x;
    if (j < E)          { src[j] = (int)ei[j]; dst[j] = (int)ei[(size_t)E + j]; }
    else if (j < E + N) { src[j] = j - E;      dst[j] = j - E; }
}
__global__ void append_loops(int* __restrict__ src, int* __restrict__ dst, int off, int N) {
    int n = blockIdx.x * blockDim.x + threadIdx.x;
    if (n < N) { src[off + n] = n; dst[off + n] = n; }
}

// ---------------- segment softmax over edges ----------------
__global__ void logits_max(const int* __restrict__ src, const int* __restrict__ dst,
                           const float* __restrict__ es, const float* __restrict__ ed,
                           float* __restrict__ logits, unsigned* __restrict__ menc, int M) {
    int e = blockIdx.x * blockDim.x + threadIdx.x;
    if (e >= M) return;
    float v = es[src[e]] + ed[dst[e]];
    v = v > 0.f ? v : ATTN_SLOPE * v;
    logits[e] = v;
    atomicMax(&menc[dst[e]], encOrd(v));
}
__global__ void z_denom(const int* __restrict__ dst, const float* __restrict__ logits,
                        const unsigned* __restrict__ menc, float* __restrict__ zb,
                        float* __restrict__ denom, int M) {
    int e = blockIdx.x * blockDim.x + threadIdx.x;
    if (e >= M) return;
    int d = dst[e];
    float z = __expf(logits[e] - decOrd(menc[d]));
    zb[e] = z;
    atomicAdd(&denom[d], z);
}
// wave-per-edge scatter: out_acc[dst] += alpha * h[k][src]
__launch_bounds__(256)
__global__ void scatter_kernel(const int* __restrict__ src, const int* __restrict__ dst,
                               const float* __restrict__ zb, const float* __restrict__ denom,
                               const float* __restrict__ hk, float* __restrict__ out_acc,
                               float* __restrict__ attn, int M) {
    int e = (blockIdx.x * blockDim.x + threadIdx.x) >> 5;
    int lane = threadIdx.x & 31;
    if (e >= M) return;
    int s = src[e], d = dst[e];
    float alpha = zb[e] / (denom[d] + 1e-16f);
    const float* hs = hk + (size_t)s * C_OUT;
    float* od = out_acc + (size_t)d * C_OUT;
    for (int c = lane; c < C_OUT; c += 32)
        atomicAdd(&od[c], alpha * hs[c]);
    if (lane == 0) attn[e] += alpha;
}

// ---------------- output activation + column L2 norm ----------------
__launch_bounds__(256)
__global__ void leaky_sumsq(const float* __restrict__ acc, float* __restrict__ xf,
                            float* __restrict__ sumsq, int total) {
    __shared__ float ss[C_OUT];
    if (threadIdx.x < C_OUT) ss[threadIdx.x] = 0.f;
    __syncthreads();
    for (int i = blockIdx.x * blockDim.x + threadIdx.x; i < total; i += gridDim.x * blockDim.x) {
        float t = acc[i] * (1.0f / K_KERNELS);
        t = t > 0.f ? t : ACT_SLOPE * t;
        xf[i] = t;
        atomicAdd(&ss[i & (C_OUT - 1)], t * t);        // ds_add_f32
    }
    __syncthreads();
    if (threadIdx.x < C_OUT) atomicAdd(&sumsq[threadIdx.x], ss[threadIdx.x]);
}
__global__ void colnorm(float* __restrict__ xf, unsigned short* __restrict__ xb,
                        const float* __restrict__ sumsq, int total) {
    for (int i = blockIdx.x * blockDim.x + threadIdx.x; i < total; i += gridDim.x * blockDim.x) {
        float nrm = fmaxf(sqrtf(sumsq[i & (C_OUT - 1)]), 1e-12f);
        float v = xf[i] / nrm;
        xf[i] = v;
        xb[i] = f2bf(v);
    }
}

// ---------------- histogram top-k edge pooling ----------------
__global__ void hist_kernel(const float* __restrict__ attn, int* __restrict__ hist, int e_cur) {
    int e = blockIdx.x * blockDim.x + threadIdx.x;
    if (e < e_cur) atomicAdd(&hist[binOf(attn[e])], 1);
}
__global__ void scan_hist(const int* __restrict__ hist, int kkeep,
                          int* __restrict__ thrBin, int* __restrict__ nAbove) {
    if (threadIdx.x == 0) {
        int acc = 0, b = 1023;
        for (; b >= 0; --b) {
            if (acc + hist[b] >= kkeep) break;
            acc += hist[b];
        }
        *thrBin = (b < 0) ? 0 : b;
        *nAbove = acc;
    }
}
__global__ void compact_kernel(const float* __restrict__ attn,
                               const int* __restrict__ src, const int* __restrict__ dst,
                               int* __restrict__ nsrc, int* __restrict__ ndst,
                               const int* __restrict__ thrBin, const int* __restrict__ nAbove,
                               int* __restrict__ cnts, int e_cur, int kkeep) {
    int e = blockIdx.x * blockDim.x + threadIdx.x;
    if (e >= e_cur) return;
    int b = binOf(attn[e]), tb = *thrBin, na = *nAbove;
    if (b > tb) {
        int pos = atomicAdd(&cnts[0], 1);
        nsrc[pos] = src[e]; ndst[pos] = dst[e];
    } else if (b == tb) {
        int t = atomicAdd(&cnts[1], 1);
        if (na + t < kkeep) { nsrc[na + t] = src[e]; ndst[na + t] = dst[e]; }
    }
}

// ---------------- outputs (written as f32 values) ----------------
__global__ void edges_out(const int* __restrict__ src, const int* __restrict__ dst,
                          float* __restrict__ out, int M) {
    int j = blockIdx.x * blockDim.x + threadIdx.x;
    if (j < M) { out[j] = (float)src[j]; out[(size_t)M + j] = (float)dst[j]; }
}

static inline int idiv(int a, int b) { return (a + b - 1) / b; }

extern "C" void kernel_launch(void* const* d_in, const int* in_sizes, int n_in,
                              void* d_out, int out_size, void* d_ws, size_t ws_size,
                              hipStream_t stream) {
    const float*     x_in  = (const float*)d_in[0];
    const float*     W0    = (const float*)d_in[1];
    const float*     Wr    = (const float*)d_in[2];
    const float*     a_src = (const float*)d_in[3];
    const float*     a_dst = (const float*)d_in[4];
    const long long* ei    = (const long long*)d_in[5];
    const int N = in_sizes[0] / DIN0;
    const int E = in_sizes[5] / 2;

    // ---- workspace layout ----
    char* base = (char*)d_ws;
    size_t cur = 0;
    auto alloc = [&](size_t bytes) -> void* {
        void* p = base + cur;
        cur = (cur + bytes + 255) & ~(size_t)255;
        return p;
    };
    unsigned short* xb   = (unsigned short*)alloc((size_t)N * DIN0 * 2);
    unsigned short* wb0  = (unsigned short*)alloc((size_t)K_KERNELS * DIN0 * C_OUT * 2);   // transposed
    unsigned short* wbr  = (unsigned short*)alloc((size_t)(DEPTH - 1) * K_KERNELS * C_OUT * C_OUT * 2);
    float* h      = (float*)alloc((size_t)K_KERNELS * N * C_OUT * 4);
    float* es     = (float*)alloc((size_t)K_KERNELS * N * 4);
    float* ed     = (float*)alloc((size_t)K_KERNELS * N * 4);
    float* xf     = (float*)alloc((size_t)N * C_OUT * 4);
    float* outAcc = (float*)alloc((size_t)N * C_OUT * 4);
    int cap = E + N;
    int* srcA = (int*)alloc((size_t)cap * 4);
    int* dstA = (int*)alloc((size_t)cap * 4);
    int* srcB = (int*)alloc((size_t)cap * 4);
    int* dstB = (int*)alloc((size_t)cap * 4);
    float* logits = (float*)alloc((size_t)cap * 4);
    float* zb     = (float*)alloc((size_t)cap * 4);
    float* attn   = (float*)alloc((size_t)cap * 4);
    unsigned* menc = (unsigned*)alloc((size_t)N * 4);
    float* denom   = (float*)alloc((size_t)N * 4);
    float* sumsq   = (float*)alloc(C_OUT * 4);
    int* hist      = (int*)alloc(1024 * 4);
    int* thrBin    = (int*)alloc(16);
    int* nAbove    = (int*)alloc(16);
    int* cnts      = (int*)alloc(16);

    float* out = (float*)d_out;

    // ---- output offsets (return order: x, latent_edge, el0, el1, el2) ----
    int k1 = E / 2;  if (k1 < 1) k1 = 1;
    int k2 = k1 / 2; if (k2 < 1) k2 = 1;
    int k3 = k2 / 2; if (k3 < 1) k3 = 1;
    size_t off_latent = (size_t)N * C_OUT;
    size_t off_el[DEPTH];
    off_el[0] = off_latent + 2 * (size_t)(k3 + N);
    off_el[1] = off_el[0] + 2 * (size_t)(E + N);
    off_el[2] = off_el[1] + 2 * (size_t)(k1 + N);

    // ---- preprocessing: bf16 conversions (weights transposed), initial edge list ----
    cvt_f32_bf16<<<idiv(N * DIN0, 256), 256, 0, stream>>>(x_in, xb, N * DIN0);
    cvt_w_transpose<<<idiv(K_KERNELS * C_OUT * DIN0, 256), 256, 0, stream>>>(W0, wb0, DIN0, K_KERNELS);
    cvt_w_transpose<<<idiv((DEPTH - 1) * K_KERNELS * C_OUT * C_OUT, 256), 256, 0, stream>>>(
        Wr, wbr, C_OUT, (DEPTH - 1) * K_KERNELS);
    init_edges<<<idiv(E + N, 256), 256, 0, stream>>>(ei, srcA, dstA, E, N);

    int e_cur = E;
    int *curS = srcA, *curD = dstA, *altS = srcB, *altD = dstB;

    for (int layer = 0; layer < DEPTH; ++layer) {
        const int M   = e_cur + N;
        const int Din = (layer == 0) ? DIN0 : C_OUT;
        const unsigned short* wb = (layer == 0) ? wb0
                                 : wbr + (size_t)(layer - 1) * K_KERNELS * C_OUT * C_OUT;

        // edge_list[layer] output
        edges_out<<<idiv(M, 256), 256, 0, stream>>>(curS, curD, out + off_el[layer], M);

        // GEMMs: h[k] = xb @ W[k]   (v_wmma_f32_16x16x32_bf16, B tile via TDM)
        dim3 ggrid(idiv(N, 128), C_OUT / COLS_PER_BLOCK, K_KERNELS);
        gemm_wmma_kernel<<<ggrid, 256, 0, stream>>>(xb, wb, h, N, Din);

        // per-node scores for this layer
        score_kernel<<<idiv(N, 8), 256, 0, stream>>>(
            h, a_src + (size_t)layer * K_KERNELS * C_OUT,
               a_dst + (size_t)layer * K_KERNELS * C_OUT, es, ed, N);

        (void)hipMemsetAsync(outAcc, 0, (size_t)N * C_OUT * 4, stream);
        (void)hipMemsetAsync(attn, 0, (size_t)M * 4, stream);

        for (int k = 0; k < K_KERNELS; ++k) {
            (void)hipMemsetAsync(menc, 0, (size_t)N * 4, stream);   // encOrd-min
            (void)hipMemsetAsync(denom, 0, (size_t)N * 4, stream);
            const float* esk = es + (size_t)k * N;
            const float* edk = ed + (size_t)k * N;
            const float* hk  = h + (size_t)k * N * C_OUT;
            logits_max<<<idiv(M, 256), 256, 0, stream>>>(curS, curD, esk, edk, logits, menc, M);
            z_denom<<<idiv(M, 256), 256, 0, stream>>>(curD, logits, menc, zb, denom, M);
            scatter_kernel<<<idiv(M, 8), 256, 0, stream>>>(curS, curD, zb, denom, hk, outAcc, attn, M);
        }

        // mean over kernels + leaky + column L2 normalize; prep bf16 x for next GEMM
        (void)hipMemsetAsync(sumsq, 0, C_OUT * 4, stream);
        leaky_sumsq<<<512, 256, 0, stream>>>(outAcc, xf, sumsq, N * C_OUT);
        colnorm<<<512, 256, 0, stream>>>(xf, xb, sumsq, N * C_OUT);

        // top-k edge pooling on attn[0:e_cur]
        int kkeep = e_cur / 2; if (kkeep < 1) kkeep = 1;
        (void)hipMemsetAsync(hist, 0, 1024 * 4, stream);
        (void)hipMemsetAsync(cnts, 0, 16, stream);
        hist_kernel<<<idiv(e_cur, 256), 256, 0, stream>>>(attn, hist, e_cur);
        scan_hist<<<1, 32, 0, stream>>>(hist, kkeep, thrBin, nAbove);
        compact_kernel<<<idiv(e_cur, 256), 256, 0, stream>>>(attn, curS, curD, altS, altD,
                                                             thrBin, nAbove, cnts, e_cur, kkeep);
        append_loops<<<idiv(N, 256), 256, 0, stream>>>(altS, altD, kkeep, N);

        int* t;
        t = curS; curS = altS; altS = t;
        t = curD; curD = altD; altD = t;
        e_cur = kkeep;
    }

    // latent_edge output
    edges_out<<<idiv(e_cur + N, 256), 256, 0, stream>>>(curS, curD, out + off_latent, e_cur + N);
    // final node features
    (void)hipMemcpyAsync(d_out, xf, (size_t)N * C_OUT * 4, hipMemcpyDeviceToDevice, stream);

    (void)n_in; (void)out_size; (void)ws_size;
}
